// DecoderBlock_18124761989660
// MI455X (gfx1250) — matrix operand order repaired and checked by reading, not compiled, not run on hardware
//
#include <hip/hip_runtime.h>
#include <hip/hip_bf16.h>

// ---------------------------------------------------------------------------
// MI455X (gfx1250) decoder block.
// Compute-bound (~340 GFLOP vs ~10us of HBM traffic at 23.3 TB/s), so every
// GEMM runs on v_wmma_f32_16x16x32_f16 (f16 inputs, f32 accumulation).
// Staging is software-pipelined: global loads for tile kt+1 are issued before
// the 16 WMMAs of tile kt; f32->f16 convert + ds_store happen after, so the
// s_wait_loadcnt lands behind a full WMMA block instead of per-load.
// ---------------------------------------------------------------------------

typedef _Float16 v16h __attribute__((ext_vector_type(16)));
typedef _Float16 v8h  __attribute__((ext_vector_type(8)));
typedef _Float16 v4h  __attribute__((ext_vector_type(4)));
typedef float    v8f  __attribute__((ext_vector_type(8)));

#define BM 128
#define BN 128
#define BK 32
#define LDS_STRIDE 40   // halfs per row: 32 + 8 pad -> 80B stride (16B aligned)

// Generic GEMM: C[row, col] = alpha * sum_k A[row,k] * B'[k,col] + bias[col]
//   transB=1: B stored [Ncols][K] row-major (weights, q@k^T)   -> B'[k,c] = B[c,k]
//   transB=0: B stored [K][Ncols] row-major (probs@V)          -> B'[k,c] = B[k,c]
// z-batching via element strides (per-head attention batching).
__global__ __launch_bounds__(128)
void gemm_wmma_f16(const float* __restrict__ A, const float* __restrict__ B,
                   float* __restrict__ C, const float* __restrict__ bias,
                   int Kd, int lda, int ldb, int ldc,
                   long sAz, long sBz, long sCz,
                   float alpha, int transB, int relu)
{
    __shared__ _Float16 As[2][BM * LDS_STRIDE];
    __shared__ _Float16 Bs[2][BN * LDS_STRIDE];

    const int tid  = threadIdx.x;
    const int lane = tid & 31;
    const int wave = tid >> 5;
    const int wm   = (wave >> 1) * 64;      // wave row offset within block
    const int wn   = (wave & 1) * 64;       // wave col offset within block

    const long z  = blockIdx.z;
    const int  n0 = blockIdx.x * BN;
    const float* Ab = A + z * sAz + (long)blockIdx.y * BM * lda;
    const float* Bb = B + z * sBz;

    // staging thread mapping
    const int rg = tid >> 3;          // 0..15 (row group, A / transB)
    const int sg = tid & 7;           // 0..7, 4 floats each
    const int kg = tid >> 5;          // 0..3  (k group, notrans B)
    const int ng = (tid & 31) * 4;    // 0..124 (col group, notrans B)

    v8f acc[4][4];
#pragma unroll
    for (int i = 0; i < 4; ++i)
#pragma unroll
        for (int j = 0; j < 4; ++j)
#pragma unroll
            for (int e = 0; e < 8; ++e) acc[i][j][e] = 0.0f;

    // ---- pipelined staging: global -> regs (issue early) ----
    float4 fa[8], fb[8];

    auto loadA = [&](int kt) {
        const float* src = Ab + (long)kt * BK;
#pragma unroll
        for (int p = 0; p < 8; ++p)
            fa[p] = *(const float4*)(src + (long)(rg + p * 16) * lda + sg * 4);
    };
    auto loadB = [&](int kt) {
        if (transB) {
            const float* src = Bb + (long)n0 * ldb + (long)kt * BK;
#pragma unroll
            for (int p = 0; p < 8; ++p)
                fb[p] = *(const float4*)(src + (long)(rg + p * 16) * ldb + sg * 4);
        } else {
            const float* src = Bb + (long)kt * BK * ldb + n0;
#pragma unroll
            for (int p = 0; p < 8; ++p)
                fb[p] = *(const float4*)(src + (long)(kg + p * 4) * ldb + ng);
        }
    };
    // ---- regs -> LDS f16 (convert late, after the WMMA block) ----
    auto storeA = [&](int buf) {
#pragma unroll
        for (int p = 0; p < 8; ++p) {
            v4h h; h[0] = (_Float16)fa[p].x; h[1] = (_Float16)fa[p].y;
                   h[2] = (_Float16)fa[p].z; h[3] = (_Float16)fa[p].w;
            *(v4h*)&As[buf][(rg + p * 16) * LDS_STRIDE + sg * 4] = h;
        }
    };
    auto storeB = [&](int buf) {
        if (transB) {
#pragma unroll
            for (int p = 0; p < 8; ++p) {
                v4h h; h[0] = (_Float16)fb[p].x; h[1] = (_Float16)fb[p].y;
                       h[2] = (_Float16)fb[p].z; h[3] = (_Float16)fb[p].w;
                *(v4h*)&Bs[buf][(rg + p * 16) * LDS_STRIDE + sg * 4] = h;
            }
        } else {   // transpose while storing
#pragma unroll
            for (int p = 0; p < 8; ++p) {
                const int k = kg + p * 4;
                Bs[buf][(ng + 0) * LDS_STRIDE + k] = (_Float16)fb[p].x;
                Bs[buf][(ng + 1) * LDS_STRIDE + k] = (_Float16)fb[p].y;
                Bs[buf][(ng + 2) * LDS_STRIDE + k] = (_Float16)fb[p].z;
                Bs[buf][(ng + 3) * LDS_STRIDE + k] = (_Float16)fb[p].w;
            }
        }
    };

    // ---- fragment loads, CDNA5 wave32 WMMA layouts ----
    auto fragA = [&](const _Float16* S, int row0) -> v16h {
        const int r  = row0 + (lane & 15);
        const int kh = (lane >> 4) * 8;     // lanes 0-15: K 0-7/16-23; 16-31: K 8-15/24-31
        const _Float16* p = S + r * LDS_STRIDE;
        v8h lo = *(const v8h*)(p + kh);
        v8h hi = *(const v8h*)(p + 16 + kh);
        return __builtin_shufflevector(lo, hi, 0,1,2,3,4,5,6,7,8,9,10,11,12,13,14,15);
    };
    auto fragB = [&](const _Float16* S, int col0) -> v16h {
        const int c  = col0 + (lane & 15);
        const int kb = (lane >> 4) * 8;     // lanes 0-15: K 0-7/16-23; 16-31: K 8-15/24-31
        const _Float16* p = S + c * LDS_STRIDE + kb;
        v8h lo = *(const v8h*)(p);
        v8h hi = *(const v8h*)(p + 16);
        return __builtin_shufflevector(lo, hi, 0,1,2,3,4,5,6,7,8,9,10,11,12,13,14,15);
    };

    const int KT = Kd / BK;
    loadA(0); loadB(0);
    storeA(0); storeB(0);
    __syncthreads();

    for (int kt = 0; kt < KT; ++kt) {
        const int buf = kt & 1;
        const bool pf = (kt + 1 < KT);

        // 1) issue next tile's global loads (in flight during the WMMAs)
        if (pf) { loadA(kt + 1); loadB(kt + 1); }

        // 2) compute on the current LDS tile
        v16h af[4], bfv[4];
#pragma unroll
        for (int i = 0; i < 4; ++i) af[i]  = fragA(As[buf], wm + i * 16);
#pragma unroll
        for (int j = 0; j < 4; ++j) bfv[j] = fragB(Bs[buf], wn + j * 16);

#pragma unroll
        for (int i = 0; i < 4; ++i)
#pragma unroll
            for (int j = 0; j < 4; ++j)
                acc[i][j] = __builtin_amdgcn_wmma_f32_16x16x32_f16(
                    false, af[i], false, bfv[j], (short)0, acc[i][j], false, false);

        // 3) convert + park next tile into the other LDS buffer
        if (pf) { storeA(buf ^ 1); storeB(buf ^ 1); }
        __syncthreads();
    }

    // ---- epilogue: C layout VGPR e -> row e (lanes 0-15) / e+8 (lanes 16-31)
    const int rbase = blockIdx.y * BM + wm + 8 * (lane >> 4);
    const int cbase = n0 + wn + (lane & 15);
#pragma unroll
    for (int j = 0; j < 4; ++j) {
        const int col = cbase + j * 16;
        const float bv = bias ? bias[col] : 0.0f;
#pragma unroll
        for (int i = 0; i < 4; ++i)
#pragma unroll
            for (int e = 0; e < 8; ++e) {
                const int row = rbase + i * 16 + e;
                float v = acc[i][j][e] * alpha + bv;
                if (relu) v = fmaxf(v, 0.0f);
                C[z * sCz + (long)row * ldc + col] = v;
            }
    }
}

// In-place row softmax (rows of `ncol` f32), one 128-thread block per row.
__global__ __launch_bounds__(128)
void softmax_rows(float* __restrict__ x, int ncol)
{
    __shared__ float red[128];
    float* row = x + (size_t)blockIdx.x * ncol;
    const int tid = threadIdx.x;

    float m = -3.4e38f;
    for (int c = tid; c < ncol; c += 128) m = fmaxf(m, row[c]);
    red[tid] = m; __syncthreads();
    for (int s = 64; s > 0; s >>= 1) {
        if (tid < s) red[tid] = fmaxf(red[tid], red[tid + s]);
        __syncthreads();
    }
    m = red[0]; __syncthreads();

    float sum = 0.0f;
    for (int c = tid; c < ncol; c += 128) {
        const float e = __expf(row[c] - m);
        row[c] = e; sum += e;
    }
    red[tid] = sum; __syncthreads();
    for (int s = 64; s > 0; s >>= 1) {
        if (tid < s) red[tid] += red[tid + s];
        __syncthreads();
    }
    const float inv = 1.0f / red[0];
    for (int c = tid; c < ncol; c += 128) row[c] *= inv;
}

// out = LN(xa + xb) * g + b, rows of 1024, 256 threads x 4 elems.
__global__ __launch_bounds__(256)
void add_layernorm(const float* __restrict__ xa, const float* __restrict__ xb,
                   const float* __restrict__ g, const float* __restrict__ b,
                   float* __restrict__ out, int ncol)
{
    __shared__ float s1[256], s2[256];
    const size_t base = (size_t)blockIdx.x * ncol;
    const int tid = threadIdx.x;

    float vals[4], sum = 0.0f, sq = 0.0f;
#pragma unroll
    for (int i = 0; i < 4; ++i) {
        const int c = tid + i * 256;
        const float v = xa[base + c] + xb[base + c];
        vals[i] = v; sum += v; sq += v * v;
    }
    s1[tid] = sum; s2[tid] = sq; __syncthreads();
    for (int s = 128; s > 0; s >>= 1) {
        if (tid < s) { s1[tid] += s1[tid + s]; s2[tid] += s2[tid + s]; }
        __syncthreads();
    }
    const float mean = s1[0] * (1.0f / 1024.0f);
    const float var  = s2[0] * (1.0f / 1024.0f) - mean * mean;
    const float inv  = rsqrtf(var + 1e-10f);
#pragma unroll
    for (int i = 0; i < 4; ++i) {
        const int c = tid + i * 256;
        out[base + c] = g[c] * (vals[i] - mean) * inv + b[c];
    }
}

// ---------------------------------------------------------------------------
static inline void gemm(hipStream_t s, const float* A, const float* B, float* C,
                        const float* bias, int Mrows, int Ncols, int Kd,
                        int lda, int ldb, int ldc,
                        long sAz, long sBz, long sCz, int Z,
                        float alpha, int transB, int relu)
{
    dim3 grid(Ncols / BN, Mrows / BM, Z), blk(128);
    gemm_wmma_f16<<<grid, blk, 0, s>>>(A, B, C, bias, Kd, lda, ldb, ldc,
                                       sAz, sBz, sCz, alpha, transB, relu);
}

extern "C" void kernel_launch(void* const* d_in, const int* in_sizes, int n_in,
                              void* d_out, int out_size, void* d_ws, size_t ws_size,
                              hipStream_t stream)
{
    const float* dec   = (const float*)d_in[0];
    const float* enc   = (const float*)d_in[1];
    const float* Wq_sa = (const float*)d_in[2];
    const float* bq_sa = (const float*)d_in[3];
    const float* Wk_sa = (const float*)d_in[4];
    const float* bk_sa = (const float*)d_in[5];
    const float* Wv_sa = (const float*)d_in[6];
    const float* bv_sa = (const float*)d_in[7];
    const float* Wo_sa = (const float*)d_in[8];
    const float* bo_sa = (const float*)d_in[9];
    const float* Wq_ca = (const float*)d_in[10];
    const float* bq_ca = (const float*)d_in[11];
    const float* Wk_ca = (const float*)d_in[12];
    const float* bk_ca = (const float*)d_in[13];
    const float* Wv_ca = (const float*)d_in[14];
    const float* bv_ca = (const float*)d_in[15];
    const float* Wo_ca = (const float*)d_in[16];
    const float* bo_ca = (const float*)d_in[17];
    const float* W1    = (const float*)d_in[18];
    const float* bf1   = (const float*)d_in[19];
    const float* W2    = (const float*)d_in[20];
    const float* bf2   = (const float*)d_in[21];
    const float* g1    = (const float*)d_in[22];
    const float* b1    = (const float*)d_in[23];
    const float* g2    = (const float*)d_in[24];
    const float* b2    = (const float*)d_in[25];
    const float* g3    = (const float*)d_in[26];
    const float* b3    = (const float*)d_in[27];

    const size_t TM = 8192ull * 1024ull;              // 8M elems (one activation)
    if (ws_size < 8 * TM * sizeof(float)) return;     // need 256 MB of scratch

    float* ws   = (float*)d_ws;
    float* q    = ws;                 // [T, H*DH]
    float* kbuf = ws + 1 * TM;
    float* vbuf = ws + 2 * TM;
    float* y    = ws + 3 * TM;        // attention context, [T, H*DH]
    float* a    = ws + 4 * TM;        // projection output / FFN output
    float* x1   = ws + 5 * TM;
    float* x2   = ws + 6 * TM;
    float* sc   = ws + 7 * TM;        // scores [H,1024,1024] (reused per n)
    float* ff1  = ws;                 // FFN hidden [8192,4096] overlays q/k/v/y (dead by then)

    const float iscale = 0.08838834764831845f;        // 1/sqrt(DH=128)
    const long  SEQ2   = 1024l * 1024l;

    auto attention = [&](const float* xq, const float* xkv,
                         const float* Wq, const float* bq, const float* Wk, const float* bk,
                         const float* Wv, const float* bv, const float* Wo, const float* bo) {
        gemm(stream, xq,  Wq, q,    bq, 8192, 1024, 1024, 1024, 1024, 1024, 0, 0, 0, 1, 1.0f, 1, 0);
        gemm(stream, xkv, Wk, kbuf, bk, 8192, 1024, 1024, 1024, 1024, 1024, 0, 0, 0, 1, 1.0f, 1, 0);
        gemm(stream, xkv, Wv, vbuf, bv, 8192, 1024, 1024, 1024, 1024, 1024, 0, 0, 0, 1, 1.0f, 1, 0);
        for (int n = 0; n < 8; ++n) {
            const long nb = (long)n * SEQ2;
            // scores[h,qi,ki] = (q . k) / sqrt(DH), heads on grid.z via 128-elem strides
            gemm(stream, q + nb, kbuf + nb, sc, nullptr, 1024, 1024, 128,
                 1024, 1024, 1024, 128, 128, SEQ2, 8, iscale, 1, 0);
            softmax_rows<<<dim3(8 * 1024), dim3(128), 0, stream>>>(sc, 1024);
            // y[qi, h*128+d] = probs @ V   (V is [K][N] -> transB=0)
            gemm(stream, sc, vbuf + nb, y + nb, nullptr, 1024, 128, 1024,
                 1024, 1024, 1024, SEQ2, 128, 128, 8, 1.0f, 0, 0);
        }
        gemm(stream, y, Wo, a, bo, 8192, 1024, 1024, 1024, 1024, 1024, 0, 0, 0, 1, 1.0f, 1, 0);
    };

    // ---- self attention + LN1 ----
    attention(dec, dec, Wq_sa, bq_sa, Wk_sa, bk_sa, Wv_sa, bv_sa, Wo_sa, bo_sa);
    add_layernorm<<<dim3(8192), dim3(256), 0, stream>>>(dec, a, g1, b1, x1, 1024);

    // ---- cross attention + LN2 ----
    attention(x1, enc, Wq_ca, bq_ca, Wk_ca, bk_ca, Wv_ca, bv_ca, Wo_ca, bo_ca);
    add_layernorm<<<dim3(8192), dim3(256), 0, stream>>>(x1, a, g2, b2, x2, 1024);

    // ---- FFN + LN3 ----
    gemm(stream, x2,  W1, ff1, bf1, 8192, 4096, 1024, 1024, 1024, 4096, 0, 0, 0, 1, 1.0f, 1, 1);
    gemm(stream, ff1, W2, a,   bf2, 8192, 1024, 4096, 4096, 4096, 1024, 0, 0, 0, 1, 1.0f, 1, 0);
    add_layernorm<<<dim3(8192), dim3(256), 0, stream>>>(x2, a, g3, b3, (float*)d_out, 1024);
}
